// MultiHeadSelfAttentionBlock_32813550142070
// MI455X (gfx1250) — matrix operand order repaired and checked by reading, not compiled, not run on hardware
//
#include <hip/hip_runtime.h>
#include <hip/hip_bf16.h>
#include <math.h>

typedef __attribute__((ext_vector_type(16))) _Float16 v16h;
typedef __attribute__((ext_vector_type(8)))  _Float16 v8h;
typedef __attribute__((ext_vector_type(8)))  float    v8f;

#define BATCH 4
#define SEQ   2048
#define DIN   384
#define DV    384
#define NH    8
#define DH    48     // per-head dim
#define DHP   64     // padded per-head dim (2 x K=32 WMMA chunks)
#define DFF   1536
#define ROWS  (BATCH*SEQ)   // 8192

#define KSTEP 64     // K-tile per staged LDS buffer
#define BROWS 64     // B-tile rows (= N block)
#define BPAD  72     // padded LDS row stride in halves (144B -> bank stride 36)

__device__ __forceinline__ v8f wmma16(v16h a, v16h b, v8f c) {
  return __builtin_amdgcn_wmma_f32_16x16x32_f16(
      /*neg_a=*/false, a, /*neg_b=*/false, b,
      /*c_mod=*/(short)0, c, /*reuse_a=*/false, /*reuse_b=*/false);
}

// Async global->LDS copy of one 16-byte chunk (CDNA5 async engine, ASYNCcnt).
// GV mode: dst VGPR = LDS byte offset, 64-bit global address in VGPR pair.
__device__ __forceinline__ void async_copy_b128(unsigned lds_off,
                                                const void* gaddr) {
  asm volatile("global_load_async_to_lds_b128 %0, %1, off"
               :: "v"(lds_off), "v"((unsigned long long)(uintptr_t)gaddr)
               : "memory");
}

// ---------------------------------------------------------------------------
// f32 -> f16 conversion
// ---------------------------------------------------------------------------
__global__ void cvt_f32_f16(const float* __restrict__ in,
                            _Float16* __restrict__ out, int n) {
  int i = blockIdx.x * blockDim.x + threadIdx.x;
  if (i < n) out[i] = (_Float16)in[i];
}

// ---------------------------------------------------------------------------
// WMMA GEMM:  C[M,Ncol] = act(A[M,K] * Bw[Ncol,K]^T) (+ Res)
// block = 128 threads (4 waves); block tile = 128 rows x 64 cols.
// Wave w owns rows [blk*128 + w*32, +32) as two 16-row A fragments.
// The shared 64x64 B tile is double-buffered in LDS via the CDNA5 async
// global->LDS engine: the next tile is issued before waiting on the current
// one (`s_wait_asynccnt 4` leaves the 4 in-flight next-tile copies pending,
// relying on per-wave in-order ASYNCcnt completion), so HBM/L2 latency
// overlaps the 16 WMMAs per K-step.
// Fragment layouts used here (A and B share one contiguous k order):
//   A (16x32): lane l = row l&15, K = k0 + (l>>4)*16 .. +15 (contiguous 32B)
//   B (32x16): lane l = col l&15, K = k0 + (l>>4)*16 .. +15 (contiguous 32B)
//   C (16x16): lane l = col l&15, rows r + 8*(l>>4)
// ---------------------------------------------------------------------------
template<int ACT, int RES, int W32, int W16>
__global__ __launch_bounds__(128) void gemm_f16_t(
    const _Float16* __restrict__ A, const _Float16* __restrict__ Bw,
    float* __restrict__ C32, _Float16* __restrict__ C16,
    const float* __restrict__ Res, int M, int Ncol, int K)
{
  const int tid = threadIdx.x;
  const int w  = tid >> 5;
  const int l  = tid & 31;
  const int ln = l & 15;
  const int hi = l >> 4;
  const int m0 = blockIdx.x * 128 + w * 32;
  const int n0 = blockIdx.y * 64;

  __shared__ _Float16 bstage[2][BROWS * BPAD];
  unsigned bs_base[2];
  bs_base[0] = (unsigned)(uintptr_t)&bstage[0][0];
  bs_base[1] = (unsigned)(uintptr_t)&bstage[1][0];

  // cooperative stage of one 64x64-half B tile: 512 chunks of 16B,
  // 4 async instructions per thread (=> ASYNCcnt 4 per wave per tile)
  auto stage = [&](int buf, int k0) {
#pragma unroll
    for (int i = 0; i < 4; ++i) {
      int c    = tid + 128 * i;
      int row  = c >> 3;
      int part = c & 7;
      const _Float16* g = Bw + (size_t)(n0 + row) * K + k0 + part * 8;
      async_copy_b128(bs_base[buf] + (unsigned)(row * BPAD + part * 8) * 2, g);
    }
  };

  v8f acc[2][4] = {};
  const _Float16* arow0 = A + (size_t)(m0 + ln) * K + hi * 16;
  const _Float16* arow1 = A + (size_t)(m0 + 16 + ln) * K + hi * 16;

  const int nsteps = K / KSTEP;
  stage(0, 0);  // prologue

  for (int it = 0; it < nsteps; ++it) {
    const int k0  = it * KSTEP;
    const int buf = it & 1;
    const bool has_next = (it + 1 < nsteps);

    if (has_next) {
      stage(buf ^ 1, k0 + KSTEP);                     // issue next tile
      asm volatile("s_wait_asynccnt 0x4" ::: "memory");  // current done, next in flight
    } else {
      asm volatile("s_wait_asynccnt 0x0" ::: "memory");
    }
    __syncthreads();  // current tile visible to all 4 waves

#pragma unroll
    for (int kk = 0; kk < 2; ++kk) {
      // A fragments (2 x 16 rows), direct from global
      v16h a[2];
      {
        v8h lo0 = *(const v8h*)(arow0 + k0 + kk * 32);
        v8h hi0 = *(const v8h*)(arow0 + k0 + kk * 32 + 8);
        v8h lo1 = *(const v8h*)(arow1 + k0 + kk * 32);
        v8h hi1 = *(const v8h*)(arow1 + k0 + kk * 32 + 8);
#pragma unroll
        for (int i = 0; i < 8; ++i) {
          a[0][i] = lo0[i]; a[0][i + 8] = hi0[i];
          a[1][i] = lo1[i]; a[1][i + 8] = hi1[i];
        }
      }
#pragma unroll
      for (int nb = 0; nb < 4; ++nb) {
        v16h bfr = *(const v16h*)
            &bstage[buf][(nb * 16 + ln) * BPAD + kk * 32 + hi * 16];
#pragma unroll
        for (int mf = 0; mf < 2; ++mf)
          acc[mf][nb] = wmma16(a[mf], bfr, acc[mf][nb]);
      }
    }
    __syncthreads();  // all reads of `buf` done before it is overwritten
  }

#pragma unroll
  for (int mf = 0; mf < 2; ++mf) {
#pragma unroll
    for (int nb = 0; nb < 4; ++nb) {
      int col = n0 + nb * 16 + ln;
#pragma unroll
      for (int r = 0; r < 8; ++r) {
        int row = m0 + mf * 16 + r + 8 * hi;
        size_t idx = (size_t)row * Ncol + col;
        float v = acc[mf][nb][r];
        if (ACT == 1) v = 0.5f * v * (1.0f + erff(v * 0.70710678118654752f));
        if (RES)      v += Res[idx];
        if (W32)      C32[idx] = v;
        if (W16)      C16[idx] = (_Float16)v;
      }
    }
  }
}

// ---------------------------------------------------------------------------
// Repack Q/K/V f32 [B,N,384] -> padded per-head f16 layouts.
//   Qh/Kh: [B*H][SEQ][64] (cols 48..63 zero), Q pre-scaled by 1/sqrt(384)
//   Vt:    [B*H][48][SEQ] (transposed so PV B-operand loads are contiguous)
// ---------------------------------------------------------------------------
__global__ void repack_qkv(const float* __restrict__ Qf, const float* __restrict__ Kf,
                           const float* __restrict__ Vf, _Float16* __restrict__ Qh,
                           _Float16* __restrict__ Kh, _Float16* __restrict__ Vt)
{
  int t  = blockIdx.x * blockDim.x + threadIdx.x;
  int c  = t & (DHP - 1);
  int n  = (t >> 6) & (SEQ - 1);
  int bh = t >> 17;
  if (bh >= BATCH * NH) return;
  int b = bh >> 3, h = bh & 7;
  size_t src = ((size_t)b * SEQ + n) * DV + h * DH + c;
  float q = 0.f, k = 0.f;
  if (c < DH) { q = Qf[src]; k = Kf[src]; }
  Qh[t] = (_Float16)(q * 0.05103103630798288f);  // 1/sqrt(384)
  Kh[t] = (_Float16)k;
  if (c < DH) Vt[((size_t)bh * DH + c) * SEQ + n] = (_Float16)Vf[src];
}

// ---------------------------------------------------------------------------
// Flash attention + per-head residual (O = Qh + softmax(QK^T/s) V).
// grid = B*H*(SEQ/64); block = 128 (4 independent waves, 16 query rows each).
// ---------------------------------------------------------------------------
__global__ __launch_bounds__(128) void attn_fused(
    const _Float16* __restrict__ Qh, const _Float16* __restrict__ Kh,
    const _Float16* __restrict__ Vt, const float* __restrict__ Qf,
    float* __restrict__ Out, _Float16* __restrict__ Of16)
{
  const int w  = threadIdx.x >> 5;
  const int l  = threadIdx.x & 31;
  const int ln = l & 15;
  const int hi = l >> 4;
  const int tiles = SEQ / 64;
  const int bh = blockIdx.x / tiles;
  const int qt = blockIdx.x % tiles;
  const int b  = bh >> 3, h = bh & 7;
  const int q0 = qt * 64 + w * 16;

  __shared__ float plds[4][16][36];  // per-wave P transpose staging

  // Q A-fragments for the two K=32 chunks of the padded head dim
  v16h qa[2];
  const _Float16* qrow = Qh + ((size_t)bh * SEQ + q0 + ln) * DHP + hi * 16;
#pragma unroll
  for (int kk = 0; kk < 2; ++kk) {
    v8h lo  = *(const v8h*)(qrow + kk * 32);
    v8h hi8 = *(const v8h*)(qrow + kk * 32 + 8);
#pragma unroll
    for (int i = 0; i < 8; ++i) { qa[kk][i] = lo[i]; qa[kk][i + 8] = hi8[i]; }
  }

  v8f o[3] = {};
  float m_i[8], l_i[8];
#pragma unroll
  for (int r = 0; r < 8; ++r) { m_i[r] = -1e30f; l_i[r] = 0.f; }

  for (int j0 = 0; j0 < SEQ; j0 += 32) {
    // scores S[16 x 32] = Q * K^T  (2 col-blocks x 2 K-chunks = 4 WMMA)
    v8f s[2] = {};
#pragma unroll
    for (int cb = 0; cb < 2; ++cb) {
      const _Float16* kb = Kh + ((size_t)bh * SEQ + j0 + cb * 16 + ln) * DHP + hi * 16;
#pragma unroll
      for (int kk = 0; kk < 2; ++kk) {
        v16h bfr = *(const v16h*)(kb + kk * 32);
        s[cb] = wmma16(qa[kk], bfr, s[cb]);
      }
    }

    // online softmax: row stats live across the 16 lanes of each half-wave
    float alpha[8], p0[8], p1[8];
#pragma unroll
    for (int r = 0; r < 8; ++r) {
      float mx = fmaxf(s[0][r], s[1][r]);
      mx = fmaxf(mx, __shfl_xor(mx, 1, 32));
      mx = fmaxf(mx, __shfl_xor(mx, 2, 32));
      mx = fmaxf(mx, __shfl_xor(mx, 4, 32));
      mx = fmaxf(mx, __shfl_xor(mx, 8, 32));
      float mn = fmaxf(m_i[r], mx);
      alpha[r] = __expf(m_i[r] - mn);
      p0[r] = __expf(s[0][r] - mn);
      p1[r] = __expf(s[1][r] - mn);
      float rs = p0[r] + p1[r];
      rs += __shfl_xor(rs, 1, 32);
      rs += __shfl_xor(rs, 2, 32);
      rs += __shfl_xor(rs, 4, 32);
      rs += __shfl_xor(rs, 8, 32);
      l_i[r] = l_i[r] * alpha[r] + rs;
      m_i[r] = mn;
    }
#pragma unroll
    for (int vb = 0; vb < 3; ++vb)
#pragma unroll
      for (int r = 0; r < 8; ++r) o[vb][r] *= alpha[r];

    // transpose P (C layout: lane=col) -> A layout (lane=row) through LDS;
    // LDS ops are in-order within a wave, so only a compiler barrier is needed
#pragma unroll
    for (int r = 0; r < 8; ++r) {
      plds[w][r + 8 * hi][ln]      = p0[r];
      plds[w][r + 8 * hi][16 + ln] = p1[r];
    }
    asm volatile("" ::: "memory");
    v16h pa;
#pragma unroll
    for (int e = 0; e < 16; ++e) {
      int col = hi * 16 + e;
      pa[e] = (_Float16)plds[w][ln][col];
    }
    asm volatile("" ::: "memory");

    // O += P * V   (3 column blocks of 16 covering d=48)
#pragma unroll
    for (int vb = 0; vb < 3; ++vb) {
      const _Float16* vrow = Vt + ((size_t)bh * DH + vb * 16 + ln) * SEQ + j0 + hi * 16;
      v16h vfr = *(const v16h*)vrow;
      o[vb] = wmma16(pa, vfr, o[vb]);
    }
  }

  // normalize, add unscaled-Q residual, store f32 out + f16 copy for FFN
#pragma unroll
  for (int vb = 0; vb < 3; ++vb) {
#pragma unroll
    for (int r = 0; r < 8; ++r) {
      int qr  = q0 + r + 8 * hi;
      int col = h * DH + vb * 16 + ln;
      size_t idx = ((size_t)b * SEQ + qr) * DV + col;
      float v = o[vb][r] / l_i[r] + Qf[idx];
      Out[idx]  = v;
      Of16[idx] = (_Float16)v;
    }
  }
}

// ---------------------------------------------------------------------------
extern "C" void kernel_launch(void* const* d_in, const int* in_sizes, int n_in,
                              void* d_out, int out_size, void* d_ws, size_t ws_size,
                              hipStream_t stream)
{
  const float* x  = (const float*)d_in[0];
  const float* y  = (const float*)d_in[1];
  const float* Wq = (const float*)d_in[2];
  const float* Wk = (const float*)d_in[3];
  const float* Wv = (const float*)d_in[4];
  const float* W1 = (const float*)d_in[5];
  const float* W2 = (const float*)d_in[6];
  float* out = (float*)d_out;

  char* ws = (char*)d_ws;
  size_t off = 0;
  auto alloc = [&](size_t bytes) -> void* {
    void* p = ws + off;
    off += (bytes + 255) & ~(size_t)255;
    return p;
  };

  _Float16* xh   = (_Float16*)alloc((size_t)ROWS * DIN * 2);
  _Float16* yh   = (_Float16*)alloc((size_t)ROWS * DIN * 2);
  _Float16* Wqh  = (_Float16*)alloc((size_t)DV  * DIN * 2);
  _Float16* Wkh  = (_Float16*)alloc((size_t)DV  * DIN * 2);
  _Float16* Wvh  = (_Float16*)alloc((size_t)DV  * DIN * 2);
  _Float16* W1h  = (_Float16*)alloc((size_t)DFF * DV  * 2);
  _Float16* W2h  = (_Float16*)alloc((size_t)DV  * DFF * 2);
  float*    Qf   = (float*)   alloc((size_t)ROWS * DV * 4);
  float*    Kf   = (float*)   alloc((size_t)ROWS * DV * 4);
  float*    Vf   = (float*)   alloc((size_t)ROWS * DV * 4);
  _Float16* Qhp  = (_Float16*)alloc((size_t)BATCH * NH * SEQ * DHP * 2);
  _Float16* Khp  = (_Float16*)alloc((size_t)BATCH * NH * SEQ * DHP * 2);
  _Float16* Vt   = (_Float16*)alloc((size_t)BATCH * NH * DH * SEQ * 2);
  _Float16* Of16 = (_Float16*)alloc((size_t)ROWS * DV * 2);
  _Float16* hid  = (_Float16*)alloc((size_t)ROWS * DFF * 2);

  // 1. convert activations + weights to f16
  cvt_f32_f16<<<(ROWS * DIN + 255) / 256, 256, 0, stream>>>(x,  xh,  ROWS * DIN);
  cvt_f32_f16<<<(ROWS * DIN + 255) / 256, 256, 0, stream>>>(y,  yh,  ROWS * DIN);
  cvt_f32_f16<<<(DV * DIN  + 255) / 256, 256, 0, stream>>>(Wq, Wqh, DV * DIN);
  cvt_f32_f16<<<(DV * DIN  + 255) / 256, 256, 0, stream>>>(Wk, Wkh, DV * DIN);
  cvt_f32_f16<<<(DV * DIN  + 255) / 256, 256, 0, stream>>>(Wv, Wvh, DV * DIN);
  cvt_f32_f16<<<(DFF * DV  + 255) / 256, 256, 0, stream>>>(W1, W1h, DFF * DV);
  cvt_f32_f16<<<(DV * DFF  + 255) / 256, 256, 0, stream>>>(W2, W2h, DV * DFF);

  dim3 blk(128);
  // 2. QKV projections (f32 outputs); block tile is 128 rows x 64 cols
  gemm_f16_t<0,0,1,0><<<dim3(ROWS/128, DV/64), blk, 0, stream>>>(xh, Wqh, Qf, nullptr, nullptr, ROWS, DV, DIN);
  gemm_f16_t<0,0,1,0><<<dim3(ROWS/128, DV/64), blk, 0, stream>>>(yh, Wkh, Kf, nullptr, nullptr, ROWS, DV, DIN);
  gemm_f16_t<0,0,1,0><<<dim3(ROWS/128, DV/64), blk, 0, stream>>>(yh, Wvh, Vf, nullptr, nullptr, ROWS, DV, DIN);

  // 3. repack into padded/transposed per-head f16 layouts
  repack_qkv<<<(BATCH * NH * SEQ * DHP) / 256, 256, 0, stream>>>(Qf, Kf, Vf, Qhp, Khp, Vt);

  // 4. flash attention + residual -> d_out (f32) and Of16 (f16)
  attn_fused<<<BATCH * NH * (SEQ / 64), 128, 0, stream>>>(Qhp, Khp, Vt, Qf, out, Of16);

  // 5. FFN: hidden = gelu(O * W1^T) (f16), then d_out += hidden * W2^T
  gemm_f16_t<1,0,0,1><<<dim3(ROWS/128, DFF/64), blk, 0, stream>>>(Of16, W1h, nullptr, hid, nullptr, ROWS, DFF, DV);
  gemm_f16_t<0,1,1,0><<<dim3(ROWS/128, DV/64),  blk, 0, stream>>>(hid,  W2h, out, nullptr, out, ROWS, DV, DFF);
}
